// EncoderRNN_57526791963152
// MI455X (gfx1250) — matrix operand order repaired and checked by reading, not compiled, not run on hardware
//
#include <hip/hip_runtime.h>

typedef __attribute__((ext_vector_type(16))) __bf16 v16bf;
typedef __attribute__((ext_vector_type(8)))  float  v8f;
typedef __attribute__((ext_vector_type(4)))  unsigned int u32x4;

constexpr int kB  = 64;
constexpr int kT  = 256;
constexpr int kH  = 2048;
constexpr int kV  = 2048;
constexpr int k4H = 4 * kH;
constexpr int kKS = 64;                 // K-step per stage
constexpr int kLD = kKS + 8;            // LDS row stride (bf16) -> bank-spread

__device__ inline v8f zero8() {
    v8f v;
#pragma unroll
    for (int i = 0; i < 8; ++i) v[i] = 0.f;
    return v;
}

// A fragment (16x32 bf16): lane holds row m=lane&15; K = kb..kb+7 and kb+16..kb+23,
// kb = (lane>>4)*8.  p -> LDS row + ksub*32 + kb (16B aligned).
__device__ inline v16bf frag_a(const __bf16* p) {
    union { v16bf v; u32x4 q[2]; } u;
    u.q[0] = *(const u32x4*)p;
    u.q[1] = *(const u32x4*)(p + 16);
    return u.v;
}

// B fragment (32x16 bf16): lane holds column n=lane&15; K = kb..kb+7 and kb+16..kb+23,
// kb = (lane>>4)*8.
__device__ inline v16bf frag_b(const __bf16* p) {
    union { v16bf v; u32x4 q[2]; } u;
    u.q[0] = *(const u32x4*)p;
    u.q[1] = *(const u32x4*)(p + 16);
    return u.v;
}

// CDNA5 async global->LDS copy (16B per lane), SADDR form: per-lane byte offset
// from a uniform 64-bit base. Tracked by ASYNCcnt.
__device__ inline void async_b128(unsigned lds_byte, unsigned gvoff_byte, const void* sbase) {
    asm volatile("global_load_async_to_lds_b128 %0, %1, %2"
                 :: "v"(lds_byte), "v"(gvoff_byte), "s"(sbase)
                 : "memory");
}
__device__ inline void wait_async0() {
    asm volatile("s_wait_asynccnt 0" ::: "memory");
}

__device__ inline float sigmoidf_(float x) { return 1.0f / (1.0f + __expf(-x)); }

// ---------------------------------------------------------------- prep ------
__global__ void cvt_f32_bf16(__bf16* __restrict__ dst, const float* __restrict__ src, int n) {
    int i = blockIdx.x * blockDim.x + threadIdx.x;
    int stride = gridDim.x * blockDim.x;
    for (; i < n; i += stride) dst[i] = (__bf16)src[i];
}

// ------------------------------------------------------------- xg GEMM ------
// xg[p, g] = sum_h Eb[tok(p), h] * Wxb[g, h] + bias[g]   (M=16384, N=8192, K=2048)
// Block tile 128x64, 8 waves of 32x32, K staged 64 at a time, async double buffer.
__global__ __launch_bounds__(256) void xgemm(
    const __bf16* __restrict__ Eb, const __bf16* __restrict__ Wxb,
    const float* __restrict__ bias, const int* __restrict__ tokens,
    float* __restrict__ xg)
{
    __shared__ __align__(16) __bf16 sA[2 * 128 * kLD];
    __shared__ __align__(16) __bf16 sB[2 * 64 * kLD];

    const int tid  = threadIdx.x;
    const int lane = tid & 31;
    const int wave = tid >> 5;
    const int m_blk = blockIdx.x * 128;
    const int n_blk = blockIdx.y * 64;

    // A: 128 rows x 64 k = 8192 bf16 -> 32 per thread (4x b128)
    const int arow = tid >> 1;
    const int aseg = (tid & 1) * 32;
    const int p    = m_blk + arow;
    const int tok  = tokens[(p & (kB - 1)) * kT + (p >> 6)];
    const unsigned aRowByte = (unsigned)((size_t)tok * kH * 2u);

    // B: 64 rows x 64 k = 4096 bf16 -> 16 per thread (2x b128)
    const int brow = tid >> 2;
    const int bseg = (tid & 3) * 16;
    const unsigned bRowByte = (unsigned)((size_t)(n_blk + brow) * kH * 2u);

    const unsigned sAoff = (unsigned)(size_t)(void*)sA;
    const unsigned sBoff = (unsigned)(size_t)(void*)sB;

    const int wm = (wave >> 1) * 32;
    const int wn = (wave & 1) * 32;
    const int mrow = lane & 15;
    const int kbA  = (lane >> 4) * 8;
    const int kbB  = (lane >> 4) * 8;

    v8f acc[2][2];
#pragma unroll
    for (int i = 0; i < 2; ++i)
#pragma unroll
        for (int j = 0; j < 2; ++j) acc[i][j] = zero8();

    auto issue = [&](int k0, int buf) {
        unsigned la = sAoff + (unsigned)((buf * 128 * kLD + arow * kLD + aseg) * 2);
        unsigned ga = aRowByte + (unsigned)((k0 + aseg) * 2);
#pragma unroll
        for (int j = 0; j < 4; ++j) async_b128(la + j * 16, ga + j * 16, Eb);
        unsigned lb = sBoff + (unsigned)((buf * 64 * kLD + brow * kLD + bseg) * 2);
        unsigned gb = bRowByte + (unsigned)((k0 + bseg) * 2);
#pragma unroll
        for (int j = 0; j < 2; ++j) async_b128(lb + j * 16, gb + j * 16, Wxb);
    };

    issue(0, 0);
    const int NIT = kH / kKS;                        // 32 stages
    for (int it = 0; it < NIT; ++it) {
        wait_async0();
        __syncthreads();
        if (it + 1 < NIT) issue((it + 1) * kKS, (it + 1) & 1);

        const __bf16* bufA = sA + (it & 1) * (128 * kLD);
        const __bf16* bufB = sB + (it & 1) * (64 * kLD);
#pragma unroll
        for (int s = 0; s < 2; ++s) {
            v16bf af[2], bfv[2];
#pragma unroll
            for (int i = 0; i < 2; ++i)
                af[i] = frag_a(bufA + (wm + i * 16 + mrow) * kLD + s * 32 + kbA);
#pragma unroll
            for (int j = 0; j < 2; ++j)
                bfv[j] = frag_b(bufB + (wn + j * 16 + mrow) * kLD + s * 32 + kbB);
#pragma unroll
            for (int i = 0; i < 2; ++i)
#pragma unroll
                for (int j = 0; j < 2; ++j)
                    acc[i][j] = __builtin_amdgcn_wmma_f32_16x16x32_bf16(
                        false, af[i], false, bfv[j], (short)0, acc[i][j], false, false);
        }
        __syncthreads();
    }

    const int cm = (lane >> 4) * 8;
    const int cn = lane & 15;
#pragma unroll
    for (int i = 0; i < 2; ++i)
#pragma unroll
        for (int j = 0; j < 2; ++j) {
            const int gm = m_blk + wm + i * 16 + cm;
            const int gn = n_blk + wn + j * 16 + cn;
            const float bv = bias[gn];
            float* outp = xg + (size_t)gm * k4H + gn;
#pragma unroll
            for (int r = 0; r < 8; ++r)
                outp[(size_t)r * k4H] = acc[i][j][r] + bv;
        }
}

// ---------------------------------------------------------- LSTM step -------
// Block owns cols [n0, n0+32) of all four gates, all 64 batch rows.
// GEMM M=64, N=128, K=2048 (async double-buffered), then full cell locally.
__global__ __launch_bounds__(256) void lstm_step(
    const __bf16* __restrict__ Whb, const __bf16* __restrict__ hprev_bf,
    const float* __restrict__ xg_t, float* __restrict__ cbuf,
    __bf16* __restrict__ hnext_bf, float* __restrict__ hs_out,
    float* __restrict__ hT_out, float* __restrict__ cT_out,
    int t, int is_last)
{
    // GEMM buffers and z-staging alias the same LDS (disjoint, barrier-separated phases)
    __shared__ __align__(16) char smem[2 * 64 * kLD * 2 + 2 * 128 * kLD * 2];
    __bf16* sA = (__bf16*)smem;                                  // 2 x 64 x kLD
    __bf16* sB = (__bf16*)(smem + 2 * 64 * kLD * 2);             // 2 x 128 x kLD
    float*  sZ = (float*)smem;                                   // 64 x 132 (after GEMM)

    const int tid  = threadIdx.x;
    const int lane = tid & 31;
    const int wave = tid >> 5;
    const int n0   = blockIdx.x * 32;

    // A: 64 rows x 64 k = 4096 bf16 -> 16 per thread (2x b128)
    const int arow = tid >> 2;
    const int aseg = (tid & 3) * 16;
    const unsigned aRowByte = (unsigned)((size_t)arow * kH * 2u);

    // B: 128 rows x 64 k = 8192 bf16 -> 32 per thread (4x b128)
    const int brow = tid >> 1;
    const int bseg = (tid & 1) * 32;
    const int bgate = brow >> 5;
    const unsigned bRowByte = (unsigned)((size_t)(bgate * kH + n0 + (brow & 31)) * kH * 2u);

    const unsigned sAoff = (unsigned)(size_t)(void*)sA;
    const unsigned sBoff = (unsigned)(size_t)(void*)sB;

    const int wm = (wave >> 1) * 16;
    const int wn = (wave & 1) * 64;
    const int mrow = lane & 15;
    const int kbA  = (lane >> 4) * 8;
    const int kbB  = (lane >> 4) * 8;

    v8f acc[4];
#pragma unroll
    for (int j = 0; j < 4; ++j) acc[j] = zero8();

    auto issue = [&](int k0, int buf) {
        unsigned la = sAoff + (unsigned)((buf * 64 * kLD + arow * kLD + aseg) * 2);
        unsigned ga = aRowByte + (unsigned)((k0 + aseg) * 2);
#pragma unroll
        for (int j = 0; j < 2; ++j) async_b128(la + j * 16, ga + j * 16, hprev_bf);
        unsigned lb = sBoff + (unsigned)((buf * 128 * kLD + brow * kLD + bseg) * 2);
        unsigned gb = bRowByte + (unsigned)((k0 + bseg) * 2);
#pragma unroll
        for (int j = 0; j < 4; ++j) async_b128(lb + j * 16, gb + j * 16, Whb);
    };

    issue(0, 0);
    const int NIT = kH / kKS;
    for (int it = 0; it < NIT; ++it) {
        wait_async0();
        __syncthreads();
        if (it + 1 < NIT) issue((it + 1) * kKS, (it + 1) & 1);

        const __bf16* bufA = sA + (it & 1) * (64 * kLD);
        const __bf16* bufB = sB + (it & 1) * (128 * kLD);
#pragma unroll
        for (int s = 0; s < 2; ++s) {
            v16bf af = frag_a(bufA + (wm + mrow) * kLD + s * 32 + kbA);
#pragma unroll
            for (int j = 0; j < 4; ++j) {
                v16bf bfv = frag_b(bufB + (wn + j * 16 + mrow) * kLD + s * 32 + kbB);
                acc[j] = __builtin_amdgcn_wmma_f32_16x16x32_bf16(
                    false, af, false, bfv, (short)0, acc[j], false, false);
            }
        }
        __syncthreads();
    }

    __syncthreads();   // GEMM buffers dead; reuse LDS as sZ

    // z -> LDS, fused with xg_t add
    const int cm = (lane >> 4) * 8;
    const int cn = lane & 15;
#pragma unroll
    for (int j = 0; j < 4; ++j) {
        const int jcol = wn + j * 16 + cn;
        const int gate = jcol >> 5;
        const int gcol = gate * kH + n0 + (jcol & 31);
#pragma unroll
        for (int r = 0; r < 8; ++r) {
            const int row = wm + cm + r;
            sZ[row * 132 + jcol] = acc[j][r] + xg_t[(size_t)row * k4H + gcol];
        }
    }
    __syncthreads();

    // LSTM cell: 64 rows x 32 cols, 8 elems/thread
    for (int e = tid; e < 64 * 32; e += 256) {
        const int row = e >> 5;
        const int col = e & 31;
        const float zi = sZ[row * 132 + col];
        const float zf = sZ[row * 132 + 32 + col];
        const float zo = sZ[row * 132 + 64 + col];
        const float zj = sZ[row * 132 + 96 + col];
        const float ig = sigmoidf_(zi);
        const float fg = sigmoidf_(zf);
        const float og = sigmoidf_(zo);
        const float jg = tanhf(zj);
        const int   gcol = n0 + col;
        const float cold = cbuf[row * kH + gcol];
        const float cnew = ig * jg + fg * cold;
        const float hnew = og * tanhf(cnew);
        cbuf[row * kH + gcol]     = cnew;
        hnext_bf[row * kH + gcol] = (__bf16)hnew;
        hs_out[((size_t)row * kT + t) * kH + gcol] = hnew;
        if (is_last) {
            hT_out[row * kH + gcol] = hnew;
            cT_out[row * kH + gcol] = cnew;
        }
    }
}

// -------------------------------------------------------------- launch ------
extern "C" void kernel_launch(void* const* d_in, const int* in_sizes, int n_in,
                              void* d_out, int out_size, void* d_ws, size_t ws_size,
                              hipStream_t stream) {
    const int*   tokens    = (const int*)d_in[0];
    const float* h0        = (const float*)d_in[1];
    const float* c0        = (const float*)d_in[2];
    const float* embedding = (const float*)d_in[3];
    const float* Wh_[4]    = { (const float*)d_in[4], (const float*)d_in[5],
                               (const float*)d_in[6], (const float*)d_in[7] };
    const float* Wx_[4]    = { (const float*)d_in[8], (const float*)d_in[9],
                               (const float*)d_in[10], (const float*)d_in[11] };
    const float* b_[4]     = { (const float*)d_in[12], (const float*)d_in[13],
                               (const float*)d_in[14], (const float*)d_in[15] };

    char* ws = (char*)d_ws;
    __bf16* Wxb  = (__bf16*)ws;                       ws += (size_t)k4H * kH * sizeof(__bf16);
    __bf16* Whb  = (__bf16*)ws;                       ws += (size_t)k4H * kH * sizeof(__bf16);
    __bf16* Eb   = (__bf16*)ws;                       ws += (size_t)kV * kH * sizeof(__bf16);
    float*  bias = (float*)ws;                        ws += (size_t)k4H * sizeof(float);
    float*  cbuf = (float*)ws;                        ws += (size_t)kB * kH * sizeof(float);
    __bf16* hbf  = (__bf16*)ws;                       ws += (size_t)2 * kB * kH * sizeof(__bf16);
    float*  xg   = (float*)ws;                        ws += (size_t)kT * kB * k4H * sizeof(float);

    float* out    = (float*)d_out;
    float* hs_out = out;
    float* hT_out = out + (size_t)kB * kT * kH;
    float* cT_out = hT_out + (size_t)kB * kH;

    const int cvtThreads = 256, cvtBlocks = 1024;
    for (int g = 0; g < 4; ++g) {
        cvt_f32_bf16<<<cvtBlocks, cvtThreads, 0, stream>>>(Whb + (size_t)g * kH * kH, Wh_[g], kH * kH);
        cvt_f32_bf16<<<cvtBlocks, cvtThreads, 0, stream>>>(Wxb + (size_t)g * kH * kH, Wx_[g], kH * kH);
        hipMemcpyAsync(bias + (size_t)g * kH, b_[g], kH * sizeof(float),
                       hipMemcpyDeviceToDevice, stream);
    }
    cvt_f32_bf16<<<cvtBlocks, cvtThreads, 0, stream>>>(Eb, embedding, kV * kH);
    cvt_f32_bf16<<<cvtBlocks, cvtThreads, 0, stream>>>(hbf, h0, kB * kH);
    hipMemcpyAsync(cbuf, c0, (size_t)kB * kH * sizeof(float), hipMemcpyDeviceToDevice, stream);

    dim3 ggrid(kB * kT / 128, k4H / 64);
    xgemm<<<ggrid, 256, 0, stream>>>(Eb, Wxb, bias, tokens, xg);

    const size_t BH = (size_t)kB * kH;
    for (int t = 0; t < kT; ++t) {
        lstm_step<<<kH / 32, 256, 0, stream>>>(
            Whb, hbf + (size_t)(t & 1) * BH, xg + (size_t)t * kB * k4H,
            cbuf, hbf + (size_t)((t + 1) & 1) * BH,
            hs_out, hT_out, cT_out, t, (t == kT - 1) ? 1 : 0);
    }
}
